// GATJK_24481313587665
// MI455X (gfx1250) — matrix operand order repaired and checked by reading, not compiled, not run on hardware
//
#include <hip/hip_runtime.h>
#include <hip/hip_bf16.h>

// ---------------------------------------------------------------------------
// Types for WMMA
// ---------------------------------------------------------------------------
typedef __attribute__((ext_vector_type(16))) __bf16 v16bf;
typedef __attribute__((ext_vector_type(8)))  float  v8f;
typedef int v4i __attribute__((vector_size(16)));   // matches builtin param type

union FragBF { uint4 q[2]; v16bf v; };

// float -> bf16 (round to nearest even)
__device__ __forceinline__ unsigned short f2bf(float f) {
    unsigned int u = __float_as_uint(f);
    unsigned int r = (u + 0x7FFFu + ((u >> 16) & 1u)) >> 16;
    return (unsigned short)r;
}

// order-preserving float <-> uint encoding (for atomicMax)
__device__ __forceinline__ unsigned int encf(float f) {
    unsigned int u = __float_as_uint(f);
    return (u >> 31) ? ~u : (u | 0x80000000u);
}
__device__ __forceinline__ float decf(unsigned int u) {
    unsigned int b = (u >> 31) ? (u & 0x7FFFFFFFu) : ~u;
    return __uint_as_float(b);
}

#define NEG_SLOPE 0.2f
#define HID 256
#define HEADS 8

// ---------------------------------------------------------------------------
// gfx1250 async global->LDS copy (16B per lane), with safe fallback.
// Builtin signature (from compiler diagnostic): first param is
// int4* in address_space(1) (global); second is the LDS destination.
// ---------------------------------------------------------------------------
#if defined(__AMDGCN__) && __has_builtin(__builtin_amdgcn_global_load_async_to_lds_b128)
#define CP128(gp, lp)                                                     \
    __builtin_amdgcn_global_load_async_to_lds_b128(                       \
        (__attribute__((address_space(1))) v4i*)(gp),                     \
        (__attribute__((address_space(3))) v4i*)(lp), 0, 0)
#if __has_builtin(__builtin_amdgcn_s_wait_asynccnt)
#define ASYNC_WAIT() __builtin_amdgcn_s_wait_asynccnt(0)
#else
#define ASYNC_WAIT() asm volatile("s_wait_asynccnt 0" ::: "memory")
#endif
#else
#define CP128(gp, lp) (*(uint4*)(lp) = *(const uint4*)(gp))
#define ASYNC_WAIT() ((void)0)
#endif

// ---------------------------------------------------------------------------
// GEMM: C[M,NcStore] = A[M,Ka](bf16) @ B[Ka,Nb](bf16) (+bias), f32 accumulate.
// A rows padded so Ka % 32 == 0; B padded so Nb % 64 == 0 -> NO guards in the
// K-loop (only a row clamp; garbage rows feed D rows that are never stored).
// Block: 256 threads (8 waves). Tile: BM=128, BN=64, BK=32.
// ---------------------------------------------------------------------------
__global__ __launch_bounds__(256)
void gemm_bf16_wmma(const unsigned short* __restrict__ A,
                    const unsigned short* __restrict__ B,
                    const float* __restrict__ bias, float* __restrict__ Cout,
                    int M, int NcStore, int Nb, int Ka) {
    constexpr int BM = 128, BK = 32;
    constexpr int LDA = 40;   // 32 + 8 pad (16B-aligned rows, breaks bank cycles)
    constexpr int LDB = 72;   // 64 + 8 pad
    __shared__ __align__(16) unsigned short sA[BM * LDA];
    __shared__ __align__(16) unsigned short sB[BK * LDB];

    const int tid  = threadIdx.x;
    const int wave = tid >> 5;
    const int lane = tid & 31;
    const int rowBase = blockIdx.y * BM;
    const int nBase   = blockIdx.x * 64;

    v8f acc[4];
    #pragma unroll
    for (int i = 0; i < 4; ++i)
        #pragma unroll
        for (int j = 0; j < 8; ++j) acc[i][j] = 0.0f;

    // staging addresses (A: 32B = half row per thread; B: 16B per thread)
    const int ar = tid >> 1, acb = (tid & 1) * 16;
    int grow = rowBase + ar; if (grow >= M) grow = M - 1;   // clamp, no branch
    const unsigned short* agp = A + (long long)grow * Ka + acb;
    unsigned short* alp = &sA[ar * LDA + acb];

    const int br = tid >> 3, bcb = (tid & 7) * 8;
    const unsigned short* bgp = B + (long long)br * Nb + nBase + bcb;
    unsigned short* blp = &sB[br * LDB + bcb];

    for (int k0 = 0; k0 < Ka; k0 += BK) {
        CP128(agp + k0, alp);
        CP128(agp + k0 + 8, alp + 8);
        CP128(bgp + (long long)k0 * Nb, blp);
        ASYNC_WAIT();
        __syncthreads();

        // A fragment: lane (m = lane&15, half = lane>>4)
        // elems 0..7 : K = half*8 + 0..7 ; elems 8..15 : K = 16 + half*8 + 0..7
        FragBF af;
        {
            int m = lane & 15, half = lane >> 4;
            const unsigned short* rowp = &sA[(wave * 16 + m) * LDA];
            af.q[0] = *(const uint4*)&rowp[half * 8];
            af.q[1] = *(const uint4*)&rowp[16 + half * 8];
        }
        // 4 B fragments + WMMA: lane holds B column N=lane&15, K in same order as A
        #pragma unroll
        for (int ns = 0; ns < 4; ++ns) {
            FragBF bf_;
            const unsigned short* bp = &sB[ns * 16 + (lane & 15)];
            unsigned short* bs = (unsigned short*)&bf_;
            #pragma unroll
            for (int s = 0; s < 8; ++s) {
                bs[s]     = bp[((lane >> 4) * 8 + s) * LDB];
                bs[s + 8] = bp[(16 + (lane >> 4) * 8 + s) * LDB];
            }
            acc[ns] = __builtin_amdgcn_wmma_f32_16x16x32_bf16(
                false, af.v, false, bf_.v, (short)0, acc[ns], false, false);
        }
        __syncthreads();
    }

    // store: D layout lane L, VGPR r -> row = r + (L>=16)*8, col = L&15
    #pragma unroll
    for (int ns = 0; ns < 4; ++ns) {
        int col   = nBase + ns * 16 + (lane & 15);
        int rbase = rowBase + wave * 16 + ((lane >> 4) * 8);
        if (col < NcStore) {
            float bv = bias ? bias[col] : 0.0f;
            #pragma unroll
            for (int r = 0; r < 8; ++r) {
                int row = rbase + r;
                if (row < M) Cout[(long long)row * NcStore + col] = acc[ns][r] + bv;
            }
        }
    }
}

// ---------------------------------------------------------------------------
// f32 [srows,scols] -> bf16 [*,dcols], zero-padded (rows beyond srows and
// cols beyond scols are zero). total = drows*dcols elements.
// ---------------------------------------------------------------------------
__global__ __launch_bounds__(256)
void pad2d_bf16(const float* __restrict__ src, unsigned short* __restrict__ dst,
                int srows, int scols, int dcols, long long total) {
    long long i = (long long)blockIdx.x * blockDim.x + threadIdx.x;
    if (i >= total) return;
    int r = (int)(i / dcols), c = (int)(i % dcols);
    float v = (r < srows && c < scols) ? src[(long long)r * scols + c] : 0.0f;
    dst[i] = f2bf(v);
}

// ---------------------------------------------------------------------------
// Per-node attention logits: es[n,h] = <h[n,h,:], a_src[h,:]> (same for ed)
// ---------------------------------------------------------------------------
__global__ __launch_bounds__(256)
void attn_logits_kernel(const float* __restrict__ h, const float* __restrict__ a_src,
                        const float* __restrict__ a_dst, float* __restrict__ es,
                        float* __restrict__ ed, int n) {
    int i = blockIdx.x * blockDim.x + threadIdx.x;
    if (i >= n * HEADS) return;
    int node = i >> 3, hd = i & 7;
    const float* hp = h + (long long)node * HID + hd * 32;
    const float* as = a_src + hd * 32;
    const float* ad = a_dst + hd * 32;
    float s = 0.0f, d = 0.0f;
    #pragma unroll
    for (int c = 0; c < 32; ++c) { float v = hp[c]; s += v * as[c]; d += v * ad[c]; }
    es[i] = s; ed[i] = d;
}

__global__ __launch_bounds__(256)
void fill_f32(float* p, float v, long long n) {
    long long i = (long long)blockIdx.x * blockDim.x + threadIdx.x;
    if (i < n) p[i] = v;
}
__global__ __launch_bounds__(256)
void fill_u32(unsigned int* p, unsigned int v, long long n) {
    long long i = (long long)blockIdx.x * blockDim.x + threadIdx.x;
    if (i < n) p[i] = v;
}

__device__ __forceinline__ void edge_nodes(const int* __restrict__ src,
                                           const int* __restrict__ dst,
                                           int e, int E, int& s, int& d) {
    if (e < E) { s = src[e]; d = dst[e]; } else { s = e - E; d = e - E; }
}

// pass 1: segment max of leaky_relu(es[src]+ed[dst]) per (dst, head)
__global__ __launch_bounds__(256)
void edge_max_kernel(const int* __restrict__ src, const int* __restrict__ dst,
                     const float* __restrict__ es, const float* __restrict__ ed,
                     unsigned int* __restrict__ menc, int E, int ET) {
    int i = blockIdx.x * blockDim.x + threadIdx.x;
    if (i >= ET * HEADS) return;
    int e = i >> 3, hd = i & 7, s, d;
    edge_nodes(src, dst, e, E, s, d);
    float v = es[s * HEADS + hd] + ed[d * HEADS + hd];
    v = v > 0.0f ? v : NEG_SLOPE * v;
    atomicMax(&menc[d * HEADS + hd], encf(v));
}

// pass 2: segment sum of exp(e - m)
__global__ __launch_bounds__(256)
void edge_expsum_kernel(const int* __restrict__ src, const int* __restrict__ dst,
                        const float* __restrict__ es, const float* __restrict__ ed,
                        const unsigned int* __restrict__ menc,
                        float* __restrict__ ssum, int E, int ET) {
    int i = blockIdx.x * blockDim.x + threadIdx.x;
    if (i >= ET * HEADS) return;
    int e = i >> 3, hd = i & 7, s, d;
    edge_nodes(src, dst, e, E, s, d);
    float v = es[s * HEADS + hd] + ed[d * HEADS + hd];
    v = v > 0.0f ? v : NEG_SLOPE * v;
    float m = decf(menc[d * HEADS + hd]);
    atomicAdd(&ssum[d * HEADS + hd], __expf(v - m));
}

// pass 3: one block per edge, 256 threads = 256 channels; scatter alpha*h[src]
__global__ __launch_bounds__(256)
void edge_agg_kernel(const int* __restrict__ src, const int* __restrict__ dst,
                     const float* __restrict__ es, const float* __restrict__ ed,
                     const unsigned int* __restrict__ menc,
                     const float* __restrict__ ssum,
                     const float* __restrict__ hlin, float* __restrict__ out,
                     int E) {
    int e = blockIdx.x;
    int c = threadIdx.x;
    int hd = c >> 5, s, d;
    edge_nodes(src, dst, e, E, s, d);
    float v = es[s * HEADS + hd] + ed[d * HEADS + hd];
    v = v > 0.0f ? v : NEG_SLOPE * v;
    float m  = decf(menc[d * HEADS + hd]);
    float sm = ssum[d * HEADS + hd];
    float alpha = __expf(v - m) / (sm + 1e-16f);
    atomicAdd(&out[(long long)d * HID + c], alpha * hlin[(long long)s * HID + c]);
}

// bias + optional ELU, in place; also emits bf16 copy for the next GEMM
__global__ __launch_bounds__(256)
void bias_act_kernel(float* __restrict__ h, const float* __restrict__ b,
                     unsigned short* __restrict__ h16, int do_elu, long long n) {
    long long i = (long long)blockIdx.x * blockDim.x + threadIdx.x;
    if (i >= n) return;
    float v = h[i] + b[(int)(i & (HID - 1))];
    if (do_elu) v = v > 0.0f ? v : (__expf(v) - 1.0f);
    h[i] = v;
    h16[i] = f2bf(v);
}

// JumpingKnowledge elementwise max -> bf16 (feeds final WMMA GEMM)
__global__ __launch_bounds__(256)
void jk_max_kernel(const float* __restrict__ h0, const float* __restrict__ h1,
                   const float* __restrict__ h2, unsigned short* __restrict__ jkb,
                   long long n) {
    long long i = (long long)blockIdx.x * blockDim.x + threadIdx.x;
    if (i >= n) return;
    jkb[i] = f2bf(fmaxf(fmaxf(h0[i], h1[i]), h2[i]));
}

// in-place log_softmax over 40-wide rows (one thread per row)
__global__ __launch_bounds__(256)
void log_softmax_kernel(float* __restrict__ logits, int n, int nc) {
    int row = blockIdx.x * blockDim.x + threadIdx.x;
    if (row >= n) return;
    float* p = logits + (long long)row * nc;
    float mx = -1e30f;
    for (int j = 0; j < nc; ++j) mx = fmaxf(mx, p[j]);
    float sum = 0.0f;
    for (int j = 0; j < nc; ++j) sum += __expf(p[j] - mx);
    float lse = mx + __logf(sum);
    for (int j = 0; j < nc; ++j) p[j] = p[j] - lse;
}

// ---------------------------------------------------------------------------
// Host launcher
// ---------------------------------------------------------------------------
extern "C" void kernel_launch(void* const* d_in, const int* in_sizes, int n_in,
                              void* d_out, int out_size, void* d_ws, size_t ws_size,
                              hipStream_t stream) {
    const float* x  = (const float*)d_in[0];
    const int*   ei = (const int*)d_in[1];
    const int NN = in_sizes[0] / 500;     // 100000
    const int EE = in_sizes[1] / 2;       // 3200000
    const int ET = EE + NN;               // + self-loops
    const int* srcv = ei;
    const int* dstv = ei + EE;

    const float* W[3]  = { (const float*)d_in[2],  (const float*)d_in[6],  (const float*)d_in[10] };
    const float* AS[3] = { (const float*)d_in[3],  (const float*)d_in[7],  (const float*)d_in[11] };
    const float* AD[3] = { (const float*)d_in[4],  (const float*)d_in[8],  (const float*)d_in[12] };
    const float* BB[3] = { (const float*)d_in[5],  (const float*)d_in[9],  (const float*)d_in[13] };
    const float* Wf = (const float*)d_in[14];
    const float* bf = (const float*)d_in[15];

    // ---- workspace layout ----
    const long long nh  = (long long)NN * HID;   // N*256
    const long long nhd = (long long)NN * HEADS; // N*8
    char* base = (char*)d_ws;
    float* h_lin = (float*)base;                       base += nh * 4;
    float* hL[3];
    hL[0] = (float*)base;                              base += nh * 4;
    hL[1] = (float*)base;                              base += nh * 4;
    hL[2] = (float*)base;                              base += nh * 4;
    float* es = (float*)base;                          base += nhd * 4;
    float* ed = (float*)base;                          base += nhd * 4;
    unsigned int* menc = (unsigned int*)base;          base += nhd * 4;
    float* ssum = (float*)base;                        base += nhd * 4;
    unsigned short* xb  = (unsigned short*)base;       base += (long long)NN * 512 * 2; // x, K padded 500->512
    unsigned short* hb  = (unsigned short*)base;       base += nh * 2;                  // bf16 activations / JK
    unsigned short* Wb0 = (unsigned short*)base;       base += (long long)512 * 256 * 2;
    unsigned short* Wb1 = (unsigned short*)base;       base += (long long)256 * 256 * 2;
    unsigned short* Wb2 = (unsigned short*)base;       base += (long long)256 * 256 * 2;
    unsigned short* Wfb = (unsigned short*)base;       base += (long long)256 * 64 * 2; // N padded 40->64

    const unsigned int ENC_NEG_INF = ~0xFF800000u; // encf(-inf)
    const dim3 blk(256);
    auto cdiv = [](long long a, long long b) { return (unsigned int)((a + b - 1) / b); };

    // ---- pre-convert GEMM operands to padded bf16 (one cheap pass) ----
    pad2d_bf16<<<cdiv((long long)NN * 512, 256), blk, 0, stream>>>(x,  xb,  NN,  500, 512, (long long)NN * 512);
    pad2d_bf16<<<cdiv(512LL * 256, 256), blk, 0, stream>>>(W[0], Wb0, 500, 256, 256, 512LL * 256);
    pad2d_bf16<<<cdiv(256LL * 256, 256), blk, 0, stream>>>(W[1], Wb1, 256, 256, 256, 256LL * 256);
    pad2d_bf16<<<cdiv(256LL * 256, 256), blk, 0, stream>>>(W[2], Wb2, 256, 256, 256, 256LL * 256);
    pad2d_bf16<<<cdiv(256LL * 64, 256),  blk, 0, stream>>>(Wf,   Wfb, 256, 40,  64,  256LL * 64);
    const unsigned short* Wb[3] = { Wb0, Wb1, Wb2 };

    for (int L = 0; L < 3; ++L) {
        const unsigned short* Ain = (L == 0) ? xb : hb;
        const int Ka = (L == 0) ? 512 : HID;

        // h_lin = Ain @ W[L]   (f32 out)
        dim3 ggrid(HID / 64, cdiv(NN, 128));
        gemm_bf16_wmma<<<ggrid, blk, 0, stream>>>(Ain, Wb[L], nullptr, h_lin, NN, HID, HID, Ka);

        // per-node logits
        attn_logits_kernel<<<cdiv(nhd, 256), blk, 0, stream>>>(h_lin, AS[L], AD[L], es, ed, NN);

        // segment softmax state
        fill_u32<<<cdiv(nhd, 256), blk, 0, stream>>>(menc, ENC_NEG_INF, nhd);
        fill_f32<<<cdiv(nhd, 256), blk, 0, stream>>>(ssum, 0.0f, nhd);

        edge_max_kernel<<<cdiv((long long)ET * HEADS, 256), blk, 0, stream>>>(
            srcv, dstv, es, ed, menc, EE, ET);
        edge_expsum_kernel<<<cdiv((long long)ET * HEADS, 256), blk, 0, stream>>>(
            srcv, dstv, es, ed, menc, ssum, EE, ET);

        // aggregate messages
        fill_f32<<<cdiv(nh, 256), blk, 0, stream>>>(hL[L], 0.0f, nh);
        edge_agg_kernel<<<dim3((unsigned int)ET), blk, 0, stream>>>(
            srcv, dstv, es, ed, menc, ssum, h_lin, hL[L], EE);

        // bias + ELU (no ELU on last conv); emit bf16 copy for next GEMM
        bias_act_kernel<<<cdiv(nh, 256), blk, 0, stream>>>(hL[L], BB[L], hb, (L < 2) ? 1 : 0, nh);
    }

    // JK max -> hb (bf16), final linear -> d_out, log_softmax in place
    jk_max_kernel<<<cdiv(nh, 256), blk, 0, stream>>>(hL[0], hL[1], hL[2], hb, nh);

    dim3 fgrid(1, cdiv(NN, 128));
    gemm_bf16_wmma<<<fgrid, blk, 0, stream>>>(hb, Wfb, bf, (float*)d_out, NN, 40, 64, HID);

    log_softmax_kernel<<<cdiv(NN, 256), blk, 0, stream>>>((float*)d_out, NN, 40);
}
